// Attention_4647154614988
// MI455X (gfx1250) — matrix operand order repaired and checked by reading, not compiled, not run on hardware
//
#include <hip/hip_runtime.h>
#include <hip/hip_bf16.h>

// ---------------------------------------------------------------------------
// Flash-attention for qkv (4, 1536, 2048) fp32, H=8, C=64, T=2048.
// S^T formulation: GEMM1 computes S^T tiles (M=key, N=query) so the softmax
// reduction over keys is per-lane (8 VGPRs) + one SWAPX16 ds_swizzle.
// Both GEMMs use v_wmma_f32_16x16x32_bf16 (wave32, CDNA5).
// exp2-domain softmax: log2(e) folded into the Q scale; raw v_exp_f32 via
// __builtin_amdgcn_exp2f (args <= 0, flush-to-zero is the desired behavior).
// Partner-half exp values exchanged as packed bf16 pairs (half the exps).
// ---------------------------------------------------------------------------

typedef __attribute__((ext_vector_type(16))) __bf16 v16bf;
typedef __attribute__((ext_vector_type(8)))  __bf16 v8bf;
typedef __attribute__((ext_vector_type(8)))  float  v8f;
typedef __attribute__((ext_vector_type(8)))  unsigned u32x8;

#define T_SEQ  2048
#define NHEAD  8
#define CH     64     // head dim
#define BLK_T  128    // query columns per workgroup (8 waves x 16)
#define SBLK   64     // key/value rows staged in LDS per step

// raw hardware exp2 (no denorm fixup): single v_exp_f32
__device__ __forceinline__ float ex2(float x) {
    return __builtin_amdgcn_exp2f(x);
}
// exchange with lane^16 partner (group-of-32 swizzle, xor=0x10, and=0x1f)
__device__ __forceinline__ float swz16(float x) {
    int i = __builtin_amdgcn_ds_swizzle(__builtin_bit_cast(int, x), 0x401f);
    return __builtin_bit_cast(float, i);
}
__device__ __forceinline__ unsigned swzu(unsigned x) {
    return (unsigned)__builtin_amdgcn_ds_swizzle((int)x, 0x401f);
}
__device__ __forceinline__ unsigned pk2(float a, float b) {
    unsigned short x = __builtin_bit_cast(unsigned short, (__bf16)a);
    unsigned short y = __builtin_bit_cast(unsigned short, (__bf16)b);
    return (unsigned)x | ((unsigned)y << 16);
}

__global__ __launch_bounds__(256, 1)
void Attention_4647154614988_kernel(const float* __restrict__ qkv,
                                    float* __restrict__ out) {
    // 32 KB staging, bf16: Q t-major, K s-major, V c-major -> all WMMA operand
    // reads are aligned ds_load_b128. Epilogue reuses it as a 64x128 f32 tile.
    __shared__ __align__(32) unsigned char smem[32768];
    __bf16* qS = (__bf16*)smem;             // BLK_T*CH  = 8192 bf16 (16 KB)
    __bf16* kS = (__bf16*)(smem + 16384);   // SBLK*CH   = 4096 bf16 ( 8 KB)
    __bf16* vS = (__bf16*)(smem + 24576);   // CH*SBLK   = 4096 bf16 ( 8 KB)
    float*  oS = (float*)smem;              // CH*BLK_T  = 8192 f32  (32 KB)

    const int tid  = threadIdx.x;
    const int lane = tid & 31;
    const int wave = tid >> 5;
    const int hi   = (lane >> 4) & 1;    // which K-half this lane carries
    const int ln   = lane & 15;          // N (matrix column) index

    const int qt = blockIdx.x;           // query tile   0..15
    const int bh = blockIdx.y;           // head index   0..31
    const int b  = bh >> 3;
    const int hh = bh & 7;
    const int t0 = qt * BLK_T;

    const size_t planeB = (size_t)3 * NHEAD * CH * T_SEQ;   // 1536*2048
    const float* qg = qkv + (size_t)b * planeB + (size_t)(hh * CH) * T_SEQ;
    const float* kg = qkv + (size_t)b * planeB + (size_t)(NHEAD * CH + hh * CH) * T_SEQ;
    const float* vg = qkv + (size_t)b * planeB + (size_t)(2 * NHEAD * CH + hh * CH) * T_SEQ;
    float*       og = out + (size_t)b * (NHEAD * CH) * T_SEQ + (size_t)(hh * CH) * T_SEQ;

    // ---- stage Q tile once; fold ch^-0.5 = 1/8 AND log2(e) (exp2 softmax)
    const float qscale = 0.125f * 1.44269504088896340736f;
    for (int i = tid; i < BLK_T * CH; i += 256) {
        int c = i >> 7, t = i & 127;                          // t fast: coalesced
        qS[t * CH + c] = (__bf16)(qg[(size_t)c * T_SEQ + t0 + t] * qscale);
    }
    __syncthreads();

    // Q B-operand (K = channel dim), per lane: n = tw+ln, 16 consecutive c's
    const int tw = wave * 16;
    v16bf qop0 = *(const v16bf*)&qS[(tw + ln) * CH + 0  + hi * 16];
    v16bf qop1 = *(const v16bf*)&qS[(tw + ln) * CH + 32 + hi * 16];

    v8f acc[4];
    v8f vzero = {0, 0, 0, 0, 0, 0, 0, 0};
#pragma unroll
    for (int ct = 0; ct < 4; ++ct) acc[ct] = vzero;
    float m_run = -INFINITY, l_run = 0.0f;

    for (int sb = 0; sb < T_SEQ / SBLK; ++sb) {
        const int s0g = sb * SBLK;
        // ---- stage K (s-major) and V (c-major) tiles, coalesced over s
        for (int i = tid; i < SBLK * CH; i += 256) {
            int c = i >> 6, s = i & 63;
            kS[s * CH + c] = (__bf16)kg[(size_t)c * T_SEQ + s0g + s];
        }
        for (int i = tid; i < SBLK * CH; i += 256) {
            int c = i >> 6, s = i & 63;
            vS[c * SBLK + s] = (__bf16)vg[(size_t)c * T_SEQ + s0g + s];
        }
        if (sb + 1 < T_SEQ / SBLK) {
            // warm L2 for the next tile (global_prefetch_b8)
            const size_t pr = (size_t)(tid >> 2) * T_SEQ + s0g + SBLK + (tid & 3) * 16;
            __builtin_prefetch(&kg[pr], 0, 0);
            __builtin_prefetch(&vg[pr], 0, 0);
        }
        __syncthreads();

#pragma unroll
        for (int ss = 0; ss < 2; ++ss) {
            const int sl = ss * 32;

            // ---- GEMM1: two 16x16 S^T tiles (M = key rows, N = queries)
            v8f st[2];
#pragma unroll
            for (int mh = 0; mh < 2; ++mh) {
                const int srow = (sl + mh * 16 + ln) * CH;
                v8bf lo0 = *(const v8bf*)&kS[srow + 0  + hi * 16];
                v8bf hh0 = *(const v8bf*)&kS[srow + 8  + hi * 16];
                v8bf lo1 = *(const v8bf*)&kS[srow + 32 + hi * 16];
                v8bf hh1 = *(const v8bf*)&kS[srow + 40 + hi * 16];
                v16bf a0 = __builtin_shufflevector(lo0, hh0, 0,1,2,3,4,5,6,7,8,9,10,11,12,13,14,15);
                v16bf a1 = __builtin_shufflevector(lo1, hh1, 0,1,2,3,4,5,6,7,8,9,10,11,12,13,14,15);
                v8f z = vzero;
                z = __builtin_amdgcn_wmma_f32_16x16x32_bf16(false, a0, false, qop0,
                                                            (short)0, z, false, false);
                z = __builtin_amdgcn_wmma_f32_16x16x32_bf16(false, a1, false, qop1,
                                                            (short)0, z, false, false);
                st[mh] = z;
            }

            // ---- online softmax (exp2 domain). Lane reduces its own 16
            //      scores; partner half merged via one scalar swizzle each.
            float mymax = fmaxf(st[0][0], st[1][0]);
#pragma unroll
            for (int r = 1; r < 8; ++r) mymax = fmaxf(mymax, fmaxf(st[0][r], st[1][r]));
            const float mt   = fmaxf(mymax, swz16(mymax));
            const float newm = fmaxf(m_run, mt);
            const float corr = ex2(m_run - newm);

            unsigned q0[4], q1[4];          // own exps, packed bf16 pairs
            float lt = 0.0f;
#pragma unroll
            for (int r = 0; r < 4; ++r) {
                float a  = ex2(st[0][2*r]   - newm);
                float bq = ex2(st[0][2*r+1] - newm);
                float c2 = ex2(st[1][2*r]   - newm);
                float d  = ex2(st[1][2*r+1] - newm);
                lt += (a + bq) + (c2 + d);
                q0[r] = pk2(a, bq);
                q1[r] = pk2(c2, d);
            }
            lt += swz16(lt);                // add partner's partial sum
            l_run = l_run * corr + lt;
            m_run = newm;
#pragma unroll
            for (int ct = 0; ct < 4; ++ct) acc[ct] = acc[ct] * corr;  // v_pk_mul_f32-able

            // ---- rebuild P^T in B-operand layout (K=32 keys, N=16 queries)
            //      packed-pair exchange with the partner half-wave
            u32x8 wv;
#pragma unroll
            for (int r = 0; r < 4; ++r) {
                unsigned s0w = swzu(q0[r]);
                unsigned s1w = swzu(q1[r]);
                wv[r]     = hi ? s1w : q0[r];
                wv[r + 4] = hi ? q1[r] : s0w;
            }
            v16bf bop = __builtin_bit_cast(v16bf, wv);

            // ---- GEMM2: acc += V(16c x 32s) * P^T(32s x 16t), 4 c-tiles
#pragma unroll
            for (int ct = 0; ct < 4; ++ct) {
                const int crow = (ct * 16 + ln) * SBLK;
                v8bf lo = *(const v8bf*)&vS[crow + sl + 0  + hi * 16];
                v8bf hg = *(const v8bf*)&vS[crow + sl + 8  + hi * 16];
                v16bf a2 = __builtin_shufflevector(lo, hg, 0,1,2,3,4,5,6,7,8,9,10,11,12,13,14,15);
                acc[ct] = __builtin_amdgcn_wmma_f32_16x16x32_bf16(false, a2, false, bop,
                                                                  (short)0, acc[ct], false, false);
            }
        }
        __syncthreads();   // protect kS/vS before next stage
    }

    // ---- epilogue: out[c, t] = acc / l, bounced through LDS so the global
    //      stores are fully coalesced over t (smem buffers are dead here).
    const float inv = 1.0f / l_run;
#pragma unroll
    for (int ct = 0; ct < 4; ++ct)
#pragma unroll
        for (int r = 0; r < 8; ++r) {
            const int c = ct * 16 + r + hi * 8;
            oS[c * BLK_T + tw + ln] = acc[ct][r] * inv;
        }
    __syncthreads();
    for (int i = tid; i < CH * BLK_T; i += 256) {
        int c = i >> 7, t = i & 127;
        og[(size_t)c * T_SEQ + t0 + t] = oS[c * BLK_T + t];
    }
}

extern "C" void kernel_launch(void* const* d_in, const int* in_sizes, int n_in,
                              void* d_out, int out_size, void* d_ws, size_t ws_size,
                              hipStream_t stream) {
    (void)in_sizes; (void)n_in; (void)d_ws; (void)ws_size; (void)out_size;
    const float* qkv = (const float*)d_in[0];
    float* out = (float*)d_out;
    dim3 grid(T_SEQ / BLK_T, 4 * NHEAD);   // 16 query tiles x 32 heads
    dim3 block(256);
    Attention_4647154614988_kernel<<<grid, block, 0, stream>>>(qkv, out);
}
